// CausalSelfAttention_76768245449397
// MI455X (gfx1250) — matrix operand order repaired and checked by reading, not compiled, not run on hardware
//
#include <hip/hip_runtime.h>
#include <stdint.h>

#define B_  4
#define S_  2048
#define D_  1024
#define H_  16
#define HD_ 64

typedef __attribute__((ext_vector_type(16))) __bf16          v16bf;
typedef __attribute__((ext_vector_type(8)))  float           v8f;
typedef __attribute__((ext_vector_type(16))) unsigned short  v16u;
typedef __attribute__((ext_vector_type(8)))  unsigned short  v8u;

static __device__ __forceinline__ unsigned short f32_bf16(float f) {
    unsigned int u = __builtin_bit_cast(unsigned int, f);
    u += 0x7FFFu + ((u >> 16) & 1u);          // round-to-nearest-even
    return (unsigned short)(u >> 16);
}
static __device__ __forceinline__ v16bf as_bf(v16u v) {
    return __builtin_bit_cast(v16bf, v);
}
static __device__ __forceinline__ v16u join8(v8u lo, v8u hi) {
    return __builtin_shufflevector(lo, hi, 0,1,2,3,4,5,6,7,8,9,10,11,12,13,14,15);
}
// A fragment (16x32 bf16): lane lm = row, halves at K = lh*8..+7 and 16+lh*8..+7
static __device__ __forceinline__ v16bf ldA(const unsigned short* row, int kk, int lh) {
    return as_bf(join8(*(const v8u*)(row + kk + lh * 8),
                       *(const v8u*)(row + kk + 16 + lh * 8)));
}
// B fragment (32x16 bf16) from [N,K] row-major: lane lm = N row, K = lh*8..+7 and 16+lh*8..+7
static __device__ __forceinline__ v16bf ldB(const unsigned short* nrow, int kk, int lh) {
    return as_bf(join8(*(const v8u*)(nrow + kk + lh * 8),
                       *(const v8u*)(nrow + kk + 16 + lh * 8)));
}
static __device__ __forceinline__ v8f wmma_bf16(v16bf a, v16bf b, v8f c) {
    return __builtin_amdgcn_wmma_f32_16x16x32_bf16(false, a, false, b, (short)0, c, false, false);
}

// ---------------- conversion kernels ----------------
__global__ void k_cvt_x(const float* __restrict__ x, unsigned short* __restrict__ xb, int n) {
    int i = blockIdx.x * 256 + threadIdx.x;
    if (i < n) xb[i] = f32_bf16(x[i]);
}

// w_qkv [D,3D] -> wT [3D,D]; fold (1/sqrt(HD))*log2(e) into Q columns.
__global__ void k_cvt_wqkvT(const float* __restrict__ w, unsigned short* __restrict__ wT) {
    int i = blockIdx.x * 256 + threadIdx.x;          // over D_*3*D_
    int kk = i / (3 * D_);
    int n  = i % (3 * D_);
    float f = w[i];
    if (n < D_) f *= 0.125f * 1.4426950408889634f;
    wT[(size_t)n * D_ + kk] = f32_bf16(f);
}

__global__ void k_cvt_woutT(const float* __restrict__ w, unsigned short* __restrict__ wT) {
    int i = blockIdx.x * 256 + threadIdx.x;          // over D_*D_
    int kk = i / D_;
    int n  = i % D_;
    wT[(size_t)n * D_ + kk] = f32_bf16(w[i]);
}

// ---------------- QKV projection GEMM ----------------
// grid (M/256, 3D/64), block 256 (8 waves). Wave computes 32(M) x 64(N).
// K-loop fully unrolled: fresh SSA fragments per step let the scheduler
// hoist next-step loads over current WMMAs with no rotation copies.
__global__ __launch_bounds__(256) void k_gemm_qkv(const unsigned short* __restrict__ xb,
                                                  const unsigned short* __restrict__ wT,
                                                  unsigned short* __restrict__ qo,
                                                  unsigned short* __restrict__ ko,
                                                  unsigned short* __restrict__ vTo) {
    const int lane = threadIdx.x & 31, wave = threadIdx.x >> 5;
    const int lm = lane & 15, lh = lane >> 4;
    const int m0 = blockIdx.x * 256 + wave * 32;
    const int n0 = blockIdx.y * 64;
    const unsigned short* a0 = xb + (size_t)(m0 + lm) * D_;
    const unsigned short* a1 = xb + (size_t)(m0 + 16 + lm) * D_;
    const unsigned short* br[4];
#pragma unroll
    for (int n = 0; n < 4; ++n) br[n] = wT + (size_t)(n0 + n * 16 + lm) * D_;

    v8f acc[2][4] = {};
#pragma unroll
    for (int kk = 0; kk < D_; kk += 32) {
        if ((kk & 127) == 0) {
            __builtin_prefetch(a0 + kk + 256, 0, 1);
            __builtin_prefetch(a1 + kk + 256, 0, 1);
        }
        v16bf A[2], Bf[4];
        A[0] = ldA(a0, kk, lh);
        A[1] = ldA(a1, kk, lh);
#pragma unroll
        for (int n = 0; n < 4; ++n) Bf[n] = ldB(br[n], kk, lh);
#pragma unroll
        for (int m = 0; m < 2; ++m)
#pragma unroll
            for (int n = 0; n < 4; ++n)
                acc[m][n] = wmma_bf16(A[m], Bf[n], acc[m][n]);
    }

    const int which = n0 >> 10;           // 0=Q 1=K 2=V (64-col tile = one head slice)
    const int h     = (n0 & 1023) >> 6;
#pragma unroll
    for (int mt = 0; mt < 2; ++mt)
#pragma unroll
        for (int n = 0; n < 4; ++n)
#pragma unroll
            for (int r = 0; r < 8; ++r) {
                int m = m0 + mt * 16 + r + lh * 8;
                int b = m >> 11, s = m & 2047;
                int bh = b * H_ + h;
                int hd = n * 16 + lm;
                unsigned short bv = f32_bf16(acc[mt][n][r]);
                if (which == 0)      qo[(size_t)bh * S_ * HD_ + (size_t)s * HD_ + hd] = bv;
                else if (which == 1) ko[(size_t)bh * S_ * HD_ + (size_t)s * HD_ + hd] = bv;
                else                 vTo[(size_t)bh * HD_ * S_ + (size_t)hd * S_ + s] = bv;
            }
}

// ---------------- flash attention ----------------
// grid (B*H, S/64), block 128 (4 waves). Wave owns a 16-row Q tile.
// K fragments double-buffered across key blocks; V loads hoisted before the
// softmax VALU section so they overlap it.
__global__ __launch_bounds__(128) void k_attn(const unsigned short* __restrict__ q,
                                              const unsigned short* __restrict__ k,
                                              const unsigned short* __restrict__ vT,
                                              unsigned short* __restrict__ ctxb) {
    __shared__ __attribute__((aligned(32))) unsigned short lds[4][16 * 40]; // pad 32->40
    const int lane = threadIdx.x & 31, wave = threadIdx.x >> 5;
    const int lm = lane & 15, lh = lane >> 4;
    const int bh = blockIdx.x;
    const int b = bh / H_, h = bh % H_;
    const int q0 = blockIdx.y * 64 + wave * 16;
    const unsigned short* qh = q  + (size_t)bh * S_ * HD_;
    const unsigned short* kh = k  + (size_t)bh * S_ * HD_;
    const unsigned short* vh = vT + (size_t)bh * HD_ * S_;

    v16bf Qa[2];
#pragma unroll
    for (int ks = 0; ks < 2; ++ks) {
        const unsigned short* qp = qh + (size_t)(q0 + lm) * HD_ + ks * 32 + lh * 8;
        Qa[ks] = as_bf(join8(*(const v8u*)qp, *(const v8u*)(qp + 16)));
    }
    v8f acc[4] = {};
    float ms[8], ls[8];
#pragma unroll
    for (int r = 0; r < 8; ++r) { ms[r] = -1e30f; ls[r] = 0.0f; }

    const int nkb = (q0 + 47) >> 5;                 // key blocks of 32 covering <= q0+15

    // K fragment: [n_sub][ks]; key row = kb*32 + n_sub*16 + lm, hd = ks*32+lh*8..+7 and ks*32+16+lh*8..+7
    v16bf Kc[2][2], Kn[2][2];
#pragma unroll
    for (int ns = 0; ns < 2; ++ns)
#pragma unroll
        for (int ks = 0; ks < 2; ++ks)
            Kc[ns][ks] = ldB(kh + (size_t)(ns * 16 + lm) * HD_, ks * 32, lh);

    for (int kb = 0; kb < nkb; ++kb) {
        const int kbn = (kb + 1 < nkb) ? kb + 1 : kb;   // clamped: uniform, in-bounds
#pragma unroll
        for (int ns = 0; ns < 2; ++ns)
#pragma unroll
            for (int ks = 0; ks < 2; ++ks)
                Kn[ns][ks] = ldB(kh + (size_t)(kbn * 32 + ns * 16 + lm) * HD_, ks * 32, lh);
        // V fragments for current block, issued before the softmax VALU work
        v16bf Vb[4];
#pragma unroll
        for (int n = 0; n < 4; ++n)
            Vb[n] = ldB(vh + (size_t)(n * 16 + lm) * S_, kb * 32, lh);

        v8f s0 = {}, s1 = {};
#pragma unroll
        for (int ks = 0; ks < 2; ++ks) {
            s0 = wmma_bf16(Qa[ks], Kc[0][ks], s0);
            s1 = wmma_bf16(Qa[ks], Kc[1][ks], s1);
        }
        const int kc0 = kb * 32 + lm, kc1 = kc0 + 16;
#pragma unroll
        for (int r = 0; r < 8; ++r) {
            const int qr = q0 + r + lh * 8;
            float a0 = (kc0 <= qr) ? s0[r] : -1e30f;  // causal mask (log2 domain)
            float a1 = (kc1 <= qr) ? s1[r] : -1e30f;
            float mx = fmaxf(a0, a1);
#pragma unroll
            for (int off = 1; off < 16; off <<= 1) mx = fmaxf(mx, __shfl_xor(mx, off, 32));
            float mn   = fmaxf(ms[r], mx);
            float p0   = __builtin_amdgcn_exp2f(a0 - mn);
            float p1   = __builtin_amdgcn_exp2f(a1 - mn);
            float corr = __builtin_amdgcn_exp2f(ms[r] - mn);
            float rs = p0 + p1;
#pragma unroll
            for (int off = 1; off < 16; off <<= 1) rs += __shfl_xor(rs, off, 32);
            ls[r] = ls[r] * corr + rs;
            ms[r] = mn;
#pragma unroll
            for (int n = 0; n < 4; ++n) acc[n][r] *= corr;
            const int row = r + lh * 8;               // C-layout -> LDS (pad to 40)
            lds[wave][row * 40 + lm]      = f32_bf16(p0);
            lds[wave][row * 40 + 16 + lm] = f32_bf16(p1);
        }
        // reload P in A-fragment layout (same-wave DS ops stay in order)
        v8u plo = *(const v8u*)&lds[wave][lm * 40 + lh * 8];
        v8u phi = *(const v8u*)&lds[wave][lm * 40 + 16 + lh * 8];
        v16bf Pa = as_bf(join8(plo, phi));
#pragma unroll
        for (int n = 0; n < 4; ++n)
            acc[n] = wmma_bf16(Pa, Vb[n], acc[n]);
#pragma unroll
        for (int ns = 0; ns < 2; ++ns)
#pragma unroll
            for (int ks = 0; ks < 2; ++ks)
                Kc[ns][ks] = Kn[ns][ks];
    }
    float inv[8];
#pragma unroll
    for (int r = 0; r < 8; ++r) inv[r] = 1.0f / ls[r];
#pragma unroll
    for (int n = 0; n < 4; ++n)
#pragma unroll
        for (int r = 0; r < 8; ++r) {
            int s = q0 + r + lh * 8;
            ctxb[((size_t)(b * S_ + s)) * D_ + h * HD_ + n * 16 + lm] = f32_bf16(acc[n][r] * inv[r]);
        }
}

// ---------------- output projection GEMM (fp32 out) ----------------
// grid (M/256, D/64), block 256 (8 waves). Wave computes 32x64, K fully unrolled.
__global__ __launch_bounds__(256) void k_gemm_out(const unsigned short* __restrict__ ctxb,
                                                  const unsigned short* __restrict__ wT,
                                                  float* __restrict__ out) {
    const int lane = threadIdx.x & 31, wave = threadIdx.x >> 5;
    const int lm = lane & 15, lh = lane >> 4;
    const int m0 = blockIdx.x * 256 + wave * 32;
    const int n0 = blockIdx.y * 64;
    const unsigned short* a0 = ctxb + (size_t)(m0 + lm) * D_;
    const unsigned short* a1 = ctxb + (size_t)(m0 + 16 + lm) * D_;
    const unsigned short* br[4];
#pragma unroll
    for (int n = 0; n < 4; ++n) br[n] = wT + (size_t)(n0 + n * 16 + lm) * D_;

    v8f acc[2][4] = {};
#pragma unroll
    for (int kk = 0; kk < D_; kk += 32) {
        if ((kk & 127) == 0) {
            __builtin_prefetch(a0 + kk + 256, 0, 1);
            __builtin_prefetch(a1 + kk + 256, 0, 1);
        }
        v16bf A[2], Bf[4];
        A[0] = ldA(a0, kk, lh);
        A[1] = ldA(a1, kk, lh);
#pragma unroll
        for (int n = 0; n < 4; ++n) Bf[n] = ldB(br[n], kk, lh);
#pragma unroll
        for (int m = 0; m < 2; ++m)
#pragma unroll
            for (int n = 0; n < 4; ++n)
                acc[m][n] = wmma_bf16(A[m], Bf[n], acc[m][n]);
    }

#pragma unroll
    for (int mt = 0; mt < 2; ++mt)
#pragma unroll
        for (int n = 0; n < 4; ++n)
#pragma unroll
            for (int r = 0; r < 8; ++r)
                out[(size_t)(m0 + mt * 16 + r + lh * 8) * D_ + n0 + n * 16 + lm] = acc[mt][n][r];
}

extern "C" void kernel_launch(void* const* d_in, const int* in_sizes, int n_in,
                              void* d_out, int out_size, void* d_ws, size_t ws_size,
                              hipStream_t stream) {
    (void)in_sizes; (void)n_in; (void)out_size; (void)ws_size;
    const float* x     = (const float*)d_in[0];
    const float* w_qkv = (const float*)d_in[1];
    const float* w_out = (const float*)d_in[2];
    float* out = (float*)d_out;

    char* ws = (char*)d_ws;
    size_t off = 0;
    auto carve = [&](size_t bytes) {
        char* p = ws + off;
        off += (bytes + 255) & ~(size_t)255;
        return p;
    };
    unsigned short* xb    = (unsigned short*)carve((size_t)B_ * S_ * D_ * 2);      // 16 MB
    unsigned short* wqkvT = (unsigned short*)carve((size_t)3 * D_ * D_ * 2);       //  6 MB
    unsigned short* woutT = (unsigned short*)carve((size_t)D_ * D_ * 2);           //  2 MB
    unsigned short* qb    = (unsigned short*)carve((size_t)B_ * H_ * S_ * HD_ * 2);// 16 MB
    unsigned short* kbuf  = (unsigned short*)carve((size_t)B_ * H_ * S_ * HD_ * 2);// 16 MB
    unsigned short* vTb   = (unsigned short*)carve((size_t)B_ * H_ * HD_ * S_ * 2);// 16 MB
    unsigned short* ctxb  = (unsigned short*)carve((size_t)B_ * S_ * D_ * 2);      // 16 MB

    k_cvt_x    <<<(B_ * S_ * D_) / 256, 256, 0, stream>>>(x, xb, B_ * S_ * D_);
    k_cvt_wqkvT<<<(D_ * 3 * D_) / 256, 256, 0, stream>>>(w_qkv, wqkvT);
    k_cvt_woutT<<<(D_ * D_) / 256, 256, 0, stream>>>(w_out, woutT);

    dim3 g1(B_ * S_ / 256, 3 * D_ / 64);
    k_gemm_qkv<<<g1, 256, 0, stream>>>(xb, wqkvT, qb, kbuf, vTb);

    dim3 g2(B_ * H_, S_ / 64);
    k_attn<<<g2, 128, 0, stream>>>(qb, kbuf, vTb, ctxb);

    dim3 g3(B_ * S_ / 256, D_ / 64);
    k_gemm_out<<<g3, 256, 0, stream>>>(ctxb, woutT, out);
}
